// MoE_28303834480970
// MI455X (gfx1250) — compile-verified, matrix operand order checked by reading, not run on hardware
//
#include <hip/hip_runtime.h>
#include <hip/hip_bf16.h>
#include <math.h>

#define T_TOKENS 16384
#define HID 768
#define INTER 2048
#define NEXP 8
#define LDSTR 40   // LDS row stride in elements for a 32-wide K chunk (16B aligned, conflict-skewed)

typedef unsigned short u16;                                   // bf16 bit-pattern storage
typedef __bf16 bf16x16 __attribute__((ext_vector_type(16)));  // WMMA A/B fragment
typedef float  f32x8   __attribute__((ext_vector_type(8)));   // WMMA C/D fragment

union FragAB { bf16x16 v; unsigned int u[8]; };

__device__ __forceinline__ u16 f2bf(float f) {
  unsigned int b = __float_as_uint(f);
  b += 0x7FFFu + ((b >> 16) & 1u);       // round-to-nearest-even
  return (u16)(b >> 16);
}

__device__ __forceinline__ float gelu_exact(float x) {
  return 0.5f * x * (1.0f + erff(x * 0.70710678118654752f));
}

// ---- CDNA5 async global->LDS copy (ASYNCcnt path), 16B per op -------------
__device__ __forceinline__ unsigned lds_addr32(const void* p) {
  // generic pointer to LDS: low 32 bits = byte offset within workgroup LDS allocation
  return (unsigned)(unsigned long long)p;
}
__device__ __forceinline__ void async_g2l_b128(const void* lds_dst, const void* gsrc) {
  asm volatile("global_load_async_to_lds_b128 %0, %1, off"
               :: "v"(lds_addr32(lds_dst)), "v"((unsigned long long)gsrc)
               : "memory");
}
__device__ __forceinline__ void wait_async0() { asm volatile("s_wait_asynccnt 0" ::: "memory"); }
__device__ __forceinline__ void wait_async4() { asm volatile("s_wait_asynccnt 4" ::: "memory"); }
__device__ __forceinline__ void wait_async6() { asm volatile("s_wait_asynccnt 6" ::: "memory"); }

// A/B 16-bit fragment per CDNA5 ISA layout (K chunk = 32, k0 = 0):
// lane = half*16 + (m or n); VGPR v: v<4 -> K = 2v + 8*half, v>=4 -> K = 16 + 2(v-4) + 8*half
__device__ __forceinline__ void ld_frag(const u16* rowptr, int half, FragAB& f) {
#pragma unroll
  for (int v = 0; v < 4; ++v)
    f.u[v] = *(const unsigned int*)(rowptr + 2 * v + 8 * half);
#pragma unroll
  for (int v = 4; v < 8; ++v)
    f.u[v] = *(const unsigned int*)(rowptr + 16 + 2 * (v - 4) + 8 * half);
}

#define WMMA_BF16(A, B, C) \
  __builtin_amdgcn_wmma_f32_16x16x32_bf16(false, (A).v, false, (B).v, (short)0, (C), false, false)

// ---------------------------------------------------------------- utilities
__global__ void k_zero_i32(int* p, int n) {
  int i = blockIdx.x * blockDim.x + threadIdx.x;
  if (i < n) p[i] = 0;
}

__global__ void k_cvt_bf16(const float* __restrict__ s, u16* __restrict__ d, int n4) {
  int i = blockIdx.x * blockDim.x + threadIdx.x;
  if (i >= n4) return;
  float4 v = ((const float4*)s)[i];
  u16 o[4] = { f2bf(v.x), f2bf(v.y), f2bf(v.z), f2bf(v.w) };
  ((uint2*)d)[i] = *(uint2*)o;
}

// ---------------------------------------------------------------- gating
__global__ __launch_bounds__(256) void k_gate(const float* __restrict__ x,
                                              const float* __restrict__ gw,
                                              int* __restrict__ cnt,
                                              int* __restrict__ list,
                                              float* __restrict__ wl) {
  int lane = threadIdx.x & 31;
  int wid  = threadIdx.x >> 5;
  int t = blockIdx.x * 8 + wid;
  const float* xr = x + (size_t)t * HID;
  float p[NEXP];
#pragma unroll
  for (int e = 0; e < NEXP; ++e) p[e] = 0.f;
  for (int k = lane; k < HID; k += 32) {
    float xv = xr[k];
#pragma unroll
    for (int e = 0; e < NEXP; ++e) p[e] += xv * gw[e * HID + k];
  }
#pragma unroll
  for (int off = 16; off > 0; off >>= 1)
#pragma unroll
    for (int e = 0; e < NEXP; ++e) p[e] += __shfl_xor(p[e], off, 32);

  if (lane == 0) {
    float m = p[0];
#pragma unroll
    for (int e = 1; e < NEXP; ++e) m = fmaxf(m, p[e]);
    float s = 0.f;
#pragma unroll
    for (int e = 0; e < NEXP; ++e) { p[e] = __expf(p[e] - m); s += p[e]; }
    float inv = 1.f / s;
#pragma unroll
    for (int e = 0; e < NEXP; ++e) p[e] *= inv;
    int i1 = 0;
#pragma unroll
    for (int e = 1; e < NEXP; ++e) if (p[e] > p[i1]) i1 = e;
    int i2 = (i1 == 0) ? 1 : 0;
#pragma unroll
    for (int e = 0; e < NEXP; ++e) if (e != i1 && p[e] > p[i2]) i2 = e;
    float r = p[i1] + p[i2];
    int pos = atomicAdd(&cnt[i1], 1);
    list[i1 * T_TOKENS + pos] = t * 2;
    wl[i1 * T_TOKENS + pos] = p[i1] / r;
    pos = atomicAdd(&cnt[i2], 1);
    list[i2 * T_TOKENS + pos] = t * 2 + 1;
    wl[i2 * T_TOKENS + pos] = p[i2] / r;
  }
}

// ---------------------------------------------------------------- routed fc1 (gathered)
__global__ __launch_bounds__(256) void k_ffn1(const u16* __restrict__ xb,
                                              const u16* __restrict__ fc1b,
                                              const float* __restrict__ bias1,
                                              const int* __restrict__ cnt,
                                              const int* __restrict__ list,
                                              u16* __restrict__ h) {
  const int e = blockIdx.z;
  const int ce = cnt[e];
  const int r0 = blockIdx.y * 128;
  if (r0 >= ce) return;
  const int n0 = blockIdx.x * 128;

  __shared__ u16 sA[2][128 * LDSTR];
  __shared__ u16 sB[2][128 * LDSTR];
  __shared__ int sIdx[128];
  const int tid = threadIdx.x;
  if (tid < 128) {
    int pos = r0 + tid;
    sIdx[tid] = (pos < ce) ? list[e * T_TOKENS + pos] : 0;
  }
  __syncthreads();
  const int lane = tid & 31, wid = tid >> 5;
  const int wr = wid & 3, wc = wid >> 2;
  const int half = lane >> 4, l16 = lane & 15;

  f32x8 acc[2][4];
  const f32x8 vz = {0.f, 0.f, 0.f, 0.f, 0.f, 0.f, 0.f, 0.f};
#pragma unroll
  for (int a = 0; a < 2; ++a)
#pragma unroll
    for (int b = 0; b < 4; ++b) acc[a][b] = vz;

  const u16* Bsrc = fc1b + ((size_t)e * INTER + n0) * HID;

  auto stage = [&](int kc, int buf) {
#pragma unroll
    for (int it = 0; it < 2; ++it) {
      int idx = it * 2048 + tid * 8;
      int row = idx >> 5, k = idx & 31;
      async_g2l_b128(&sA[buf][row * LDSTR + k], xb + (size_t)(sIdx[row] >> 1) * HID + kc + k);
      async_g2l_b128(&sB[buf][row * LDSTR + k], Bsrc + (size_t)row * HID + kc + k);
    }
  };

  const int NCH = HID / 32;
  stage(0, 0);
  for (int i = 0; i < NCH; ++i) {
    int cur = i & 1;
    if (i + 1 < NCH) { stage((i + 1) * 32, cur ^ 1); wait_async4(); }
    else             { wait_async0(); }
    __syncthreads();

    FragAB a0, a1, b0, b1, b2, b3;
    ld_frag(&sA[cur][(wr * 32 + l16) * LDSTR], half, a0);
    ld_frag(&sA[cur][(wr * 32 + 16 + l16) * LDSTR], half, a1);
    ld_frag(&sB[cur][(wc * 64 + 0 + l16) * LDSTR], half, b0);
    ld_frag(&sB[cur][(wc * 64 + 16 + l16) * LDSTR], half, b1);
    ld_frag(&sB[cur][(wc * 64 + 32 + l16) * LDSTR], half, b2);
    ld_frag(&sB[cur][(wc * 64 + 48 + l16) * LDSTR], half, b3);
    acc[0][0] = WMMA_BF16(a0, b0, acc[0][0]);
    acc[1][0] = WMMA_BF16(a1, b0, acc[1][0]);
    acc[0][1] = WMMA_BF16(a0, b1, acc[0][1]);
    acc[1][1] = WMMA_BF16(a1, b1, acc[1][1]);
    acc[0][2] = WMMA_BF16(a0, b2, acc[0][2]);
    acc[1][2] = WMMA_BF16(a1, b2, acc[1][2]);
    acc[0][3] = WMMA_BF16(a0, b3, acc[0][3]);
    acc[1][3] = WMMA_BF16(a1, b3, acc[1][3]);
    __syncthreads();
  }
#pragma unroll
  for (int tm = 0; tm < 2; ++tm)
#pragma unroll
    for (int tn = 0; tn < 4; ++tn)
#pragma unroll
      for (int i = 0; i < 8; ++i) {
        int rl = wr * 32 + tm * 16 + i + 8 * half;
        if (r0 + rl < ce) {
          int n = n0 + wc * 64 + tn * 16 + l16;
          float v = acc[tm][tn][i] + bias1[e * INTER + n];
          h[(size_t)sIdx[rl] * INTER + n] = f2bf(gelu_exact(v));
        }
      }
}

// ---------------------------------------------------------------- routed fc2 (+weighted scatter)
__global__ __launch_bounds__(256) void k_ffn2(const u16* __restrict__ h,
                                              const u16* __restrict__ fc2b,
                                              const float* __restrict__ bias2,
                                              const int* __restrict__ cnt,
                                              const int* __restrict__ list,
                                              const float* __restrict__ wl,
                                              float* __restrict__ out) {
  const int e = blockIdx.z;
  const int ce = cnt[e];
  const int r0 = blockIdx.y * 128;
  if (r0 >= ce) return;
  const int n0 = blockIdx.x * 128;

  __shared__ u16 sA[2][128 * LDSTR];
  __shared__ u16 sB[2][128 * LDSTR];
  __shared__ int sIdx[128];
  __shared__ float sW[128];
  const int tid = threadIdx.x;
  if (tid < 128) {
    int pos = r0 + tid;
    sIdx[tid] = (pos < ce) ? list[e * T_TOKENS + pos] : 0;
    sW[tid]   = (pos < ce) ? wl[e * T_TOKENS + pos] : 0.f;
  }
  __syncthreads();
  const int lane = tid & 31, wid = tid >> 5;
  const int wr = wid & 3, wc = wid >> 2;
  const int half = lane >> 4, l16 = lane & 15;

  f32x8 acc[2][4];
  const f32x8 vz = {0.f, 0.f, 0.f, 0.f, 0.f, 0.f, 0.f, 0.f};
#pragma unroll
  for (int a = 0; a < 2; ++a)
#pragma unroll
    for (int b = 0; b < 4; ++b) acc[a][b] = vz;

  const u16* Bsrc = fc2b + ((size_t)e * HID + n0) * INTER;

  auto stage = [&](int kc, int buf) {
#pragma unroll
    for (int it = 0; it < 2; ++it) {
      int idx = it * 2048 + tid * 8;
      int row = idx >> 5, k = idx & 31;
      async_g2l_b128(&sA[buf][row * LDSTR + k], h + (size_t)sIdx[row] * INTER + kc + k);
      async_g2l_b128(&sB[buf][row * LDSTR + k], Bsrc + (size_t)row * INTER + kc + k);
    }
  };

  const int NCH = INTER / 32;
  stage(0, 0);
  for (int i = 0; i < NCH; ++i) {
    int cur = i & 1;
    if (i + 1 < NCH) { stage((i + 1) * 32, cur ^ 1); wait_async4(); }
    else             { wait_async0(); }
    __syncthreads();

    FragAB a0, a1, b0, b1, b2, b3;
    ld_frag(&sA[cur][(wr * 32 + l16) * LDSTR], half, a0);
    ld_frag(&sA[cur][(wr * 32 + 16 + l16) * LDSTR], half, a1);
    ld_frag(&sB[cur][(wc * 64 + 0 + l16) * LDSTR], half, b0);
    ld_frag(&sB[cur][(wc * 64 + 16 + l16) * LDSTR], half, b1);
    ld_frag(&sB[cur][(wc * 64 + 32 + l16) * LDSTR], half, b2);
    ld_frag(&sB[cur][(wc * 64 + 48 + l16) * LDSTR], half, b3);
    acc[0][0] = WMMA_BF16(a0, b0, acc[0][0]);
    acc[1][0] = WMMA_BF16(a1, b0, acc[1][0]);
    acc[0][1] = WMMA_BF16(a0, b1, acc[0][1]);
    acc[1][1] = WMMA_BF16(a1, b1, acc[1][1]);
    acc[0][2] = WMMA_BF16(a0, b2, acc[0][2]);
    acc[1][2] = WMMA_BF16(a1, b2, acc[1][2]);
    acc[0][3] = WMMA_BF16(a0, b3, acc[0][3]);
    acc[1][3] = WMMA_BF16(a1, b3, acc[1][3]);
    __syncthreads();
  }
#pragma unroll
  for (int tm = 0; tm < 2; ++tm)
#pragma unroll
    for (int tn = 0; tn < 4; ++tn)
#pragma unroll
      for (int i = 0; i < 8; ++i) {
        int rl = wr * 32 + tm * 16 + i + 8 * half;
        if (r0 + rl < ce) {
          int col = n0 + wc * 64 + tn * 16 + l16;
          int token = sIdx[rl] >> 1;
          float v = sW[rl] * (acc[tm][tn][i] + bias2[e * HID + col]);
          atomicAdd(out + (size_t)token * HID + col, v);
        }
      }
}

// ---------------------------------------------------------------- shared MLP stage 1
__global__ __launch_bounds__(256) void k_shared1(const u16* __restrict__ xb,
                                                 const u16* __restrict__ w1b, const float* __restrict__ b1,
                                                 const u16* __restrict__ w3b, const float* __restrict__ b3,
                                                 u16* __restrict__ u) {
  const int r0 = blockIdx.y * 128;
  const int n0 = blockIdx.x * 128;

  __shared__ u16 sA[2][128 * LDSTR];
  __shared__ u16 sB1[2][128 * LDSTR];
  __shared__ u16 sB3[2][128 * LDSTR];
  const int tid = threadIdx.x;
  const int lane = tid & 31, wid = tid >> 5;
  const int wr = wid & 3, wc = wid >> 2;
  const int half = lane >> 4, l16 = lane & 15;

  f32x8 acc1[2][4], acc3[2][4];
  const f32x8 vz = {0.f, 0.f, 0.f, 0.f, 0.f, 0.f, 0.f, 0.f};
#pragma unroll
  for (int a = 0; a < 2; ++a)
#pragma unroll
    for (int b = 0; b < 4; ++b) { acc1[a][b] = vz; acc3[a][b] = vz; }

  const u16* B1src = w1b + (size_t)n0 * HID;
  const u16* B3src = w3b + (size_t)n0 * HID;

  auto stage = [&](int kc, int buf) {
#pragma unroll
    for (int it = 0; it < 2; ++it) {
      int idx = it * 2048 + tid * 8;
      int row = idx >> 5, k = idx & 31;
      async_g2l_b128(&sA[buf][row * LDSTR + k],  xb + (size_t)(r0 + row) * HID + kc + k);
      async_g2l_b128(&sB1[buf][row * LDSTR + k], B1src + (size_t)row * HID + kc + k);
      async_g2l_b128(&sB3[buf][row * LDSTR + k], B3src + (size_t)row * HID + kc + k);
    }
  };

  const int NCH = HID / 32;
  stage(0, 0);
  for (int i = 0; i < NCH; ++i) {
    int cur = i & 1;
    if (i + 1 < NCH) { stage((i + 1) * 32, cur ^ 1); wait_async6(); }
    else             { wait_async0(); }
    __syncthreads();

    FragAB a0, a1, b0, b1, b2, b3;
    ld_frag(&sA[cur][(wr * 32 + l16) * LDSTR], half, a0);
    ld_frag(&sA[cur][(wr * 32 + 16 + l16) * LDSTR], half, a1);
    ld_frag(&sB1[cur][(wc * 64 + 0 + l16) * LDSTR], half, b0);
    ld_frag(&sB1[cur][(wc * 64 + 16 + l16) * LDSTR], half, b1);
    ld_frag(&sB1[cur][(wc * 64 + 32 + l16) * LDSTR], half, b2);
    ld_frag(&sB1[cur][(wc * 64 + 48 + l16) * LDSTR], half, b3);
    acc1[0][0] = WMMA_BF16(a0, b0, acc1[0][0]);
    acc1[1][0] = WMMA_BF16(a1, b0, acc1[1][0]);
    acc1[0][1] = WMMA_BF16(a0, b1, acc1[0][1]);
    acc1[1][1] = WMMA_BF16(a1, b1, acc1[1][1]);
    acc1[0][2] = WMMA_BF16(a0, b2, acc1[0][2]);
    acc1[1][2] = WMMA_BF16(a1, b2, acc1[1][2]);
    acc1[0][3] = WMMA_BF16(a0, b3, acc1[0][3]);
    acc1[1][3] = WMMA_BF16(a1, b3, acc1[1][3]);
    ld_frag(&sB3[cur][(wc * 64 + 0 + l16) * LDSTR], half, b0);
    ld_frag(&sB3[cur][(wc * 64 + 16 + l16) * LDSTR], half, b1);
    ld_frag(&sB3[cur][(wc * 64 + 32 + l16) * LDSTR], half, b2);
    ld_frag(&sB3[cur][(wc * 64 + 48 + l16) * LDSTR], half, b3);
    acc3[0][0] = WMMA_BF16(a0, b0, acc3[0][0]);
    acc3[1][0] = WMMA_BF16(a1, b0, acc3[1][0]);
    acc3[0][1] = WMMA_BF16(a0, b1, acc3[0][1]);
    acc3[1][1] = WMMA_BF16(a1, b1, acc3[1][1]);
    acc3[0][2] = WMMA_BF16(a0, b2, acc3[0][2]);
    acc3[1][2] = WMMA_BF16(a1, b2, acc3[1][2]);
    acc3[0][3] = WMMA_BF16(a0, b3, acc3[0][3]);
    acc3[1][3] = WMMA_BF16(a1, b3, acc3[1][3]);
    __syncthreads();
  }
#pragma unroll
  for (int tm = 0; tm < 2; ++tm)
#pragma unroll
    for (int tn = 0; tn < 4; ++tn)
#pragma unroll
      for (int i = 0; i < 8; ++i) {
        int rl = wr * 32 + tm * 16 + i + 8 * half;
        int n = n0 + wc * 64 + tn * 16 + l16;
        float h1 = gelu_exact(acc1[tm][tn][i] + b1[n]);
        float h3 = acc3[tm][tn][i] + b3[n];
        u[(size_t)(r0 + rl) * INTER + n] = f2bf(h1 * h3);
      }
}

// ---------------------------------------------------------------- shared MLP stage 2
__global__ __launch_bounds__(256) void k_shared2(const u16* __restrict__ u,
                                                 const u16* __restrict__ w2b,
                                                 const float* __restrict__ b2,
                                                 float* __restrict__ out) {
  const int r0 = blockIdx.y * 128;
  const int n0 = blockIdx.x * 128;

  __shared__ u16 sA[2][128 * LDSTR];
  __shared__ u16 sB[2][128 * LDSTR];
  const int tid = threadIdx.x;
  const int lane = tid & 31, wid = tid >> 5;
  const int wr = wid & 3, wc = wid >> 2;
  const int half = lane >> 4, l16 = lane & 15;

  f32x8 acc[2][4];
  const f32x8 vz = {0.f, 0.f, 0.f, 0.f, 0.f, 0.f, 0.f, 0.f};
#pragma unroll
  for (int a = 0; a < 2; ++a)
#pragma unroll
    for (int b = 0; b < 4; ++b) acc[a][b] = vz;

  const u16* Bsrc = w2b + (size_t)n0 * INTER;

  auto stage = [&](int kc, int buf) {
#pragma unroll
    for (int it = 0; it < 2; ++it) {
      int idx = it * 2048 + tid * 8;
      int row = idx >> 5, k = idx & 31;
      async_g2l_b128(&sA[buf][row * LDSTR + k], u + (size_t)(r0 + row) * INTER + kc + k);
      async_g2l_b128(&sB[buf][row * LDSTR + k], Bsrc + (size_t)row * INTER + kc + k);
    }
  };

  const int NCH = INTER / 32;
  stage(0, 0);
  for (int i = 0; i < NCH; ++i) {
    int cur = i & 1;
    if (i + 1 < NCH) { stage((i + 1) * 32, cur ^ 1); wait_async4(); }
    else             { wait_async0(); }
    __syncthreads();

    FragAB a0, a1, b0, b1, b2, b3;
    ld_frag(&sA[cur][(wr * 32 + l16) * LDSTR], half, a0);
    ld_frag(&sA[cur][(wr * 32 + 16 + l16) * LDSTR], half, a1);
    ld_frag(&sB[cur][(wc * 64 + 0 + l16) * LDSTR], half, b0);
    ld_frag(&sB[cur][(wc * 64 + 16 + l16) * LDSTR], half, b1);
    ld_frag(&sB[cur][(wc * 64 + 32 + l16) * LDSTR], half, b2);
    ld_frag(&sB[cur][(wc * 64 + 48 + l16) * LDSTR], half, b3);
    acc[0][0] = WMMA_BF16(a0, b0, acc[0][0]);
    acc[1][0] = WMMA_BF16(a1, b0, acc[1][0]);
    acc[0][1] = WMMA_BF16(a0, b1, acc[0][1]);
    acc[1][1] = WMMA_BF16(a1, b1, acc[1][1]);
    acc[0][2] = WMMA_BF16(a0, b2, acc[0][2]);
    acc[1][2] = WMMA_BF16(a1, b2, acc[1][2]);
    acc[0][3] = WMMA_BF16(a0, b3, acc[0][3]);
    acc[1][3] = WMMA_BF16(a1, b3, acc[1][3]);
    __syncthreads();
  }
#pragma unroll
  for (int tm = 0; tm < 2; ++tm)
#pragma unroll
    for (int tn = 0; tn < 4; ++tn)
#pragma unroll
      for (int i = 0; i < 8; ++i) {
        int rl = wr * 32 + tm * 16 + i + 8 * half;
        int col = n0 + wc * 64 + tn * 16 + l16;
        out[(size_t)(r0 + rl) * HID + col] = acc[tm][tn][i] + b2[col];
      }
}

// ---------------------------------------------------------------- launcher
extern "C" void kernel_launch(void* const* d_in, const int* in_sizes, int n_in,
                              void* d_out, int out_size, void* d_ws, size_t ws_size,
                              hipStream_t stream) {
  (void)in_sizes; (void)n_in; (void)out_size; (void)ws_size;
  const float* x      = (const float*)d_in[0];
  const float* gate_w = (const float*)d_in[1];
  const float* fc1_w  = (const float*)d_in[2];
  const float* fc1_b  = (const float*)d_in[3];
  const float* fc2_w  = (const float*)d_in[4];
  const float* fc2_b  = (const float*)d_in[5];
  const float* w1_w   = (const float*)d_in[6];
  const float* w1_b   = (const float*)d_in[7];
  const float* w3_w   = (const float*)d_in[8];
  const float* w3_b   = (const float*)d_in[9];
  const float* w2_w   = (const float*)d_in[10];
  const float* w2_b   = (const float*)d_in[11];
  float* out = (float*)d_out;

  char* ws = (char*)d_ws;
  size_t o = 0;
  u16* xb   = (u16*)(ws + o); o += (size_t)T_TOKENS * HID * 2;
  u16* fc1b = (u16*)(ws + o); o += (size_t)NEXP * INTER * HID * 2;
  u16* fc2b = (u16*)(ws + o); o += (size_t)NEXP * HID * INTER * 2;
  u16* w1b  = (u16*)(ws + o); o += (size_t)INTER * HID * 2;
  u16* w3b  = (u16*)(ws + o); o += (size_t)INTER * HID * 2;
  u16* w2b  = (u16*)(ws + o); o += (size_t)HID * INTER * 2;
  int* cnt  = (int*)(ws + o); o += 256;
  int* list = (int*)(ws + o); o += (size_t)NEXP * T_TOKENS * 4;
  float* wl = (float*)(ws + o); o += (size_t)NEXP * T_TOKENS * 4;
  u16* big  = (u16*)(ws + o);  // u (T x INTER) then reused as h (2T x INTER)

  k_zero_i32<<<1, 64, 0, stream>>>(cnt, 64);

  auto cvt = [&](const float* s, u16* d, size_t n) {
    int n4 = (int)(n / 4);
    k_cvt_bf16<<<(n4 + 255) / 256, 256, 0, stream>>>(s, d, n4);
  };
  cvt(x,     xb,   (size_t)T_TOKENS * HID);
  cvt(fc1_w, fc1b, (size_t)NEXP * INTER * HID);
  cvt(fc2_w, fc2b, (size_t)NEXP * HID * INTER);
  cvt(w1_w,  w1b,  (size_t)INTER * HID);
  cvt(w3_w,  w3b,  (size_t)INTER * HID);
  cvt(w2_w,  w2b,  (size_t)HID * INTER);

  k_gate<<<T_TOKENS / 8, 256, 0, stream>>>(x, gate_w, cnt, list, wl);

  // shared MLP first: S2 writes `out` plainly, routed path then atomically accumulates.
  k_shared1<<<dim3(INTER / 128, T_TOKENS / 128, 1), 256, 0, stream>>>(xb, w1b, w1_b, w3b, w3_b, big);
  k_shared2<<<dim3(HID / 128, T_TOKENS / 128, 1), 256, 0, stream>>>(big, w2b, w2_b, out);

  // routed experts: `big` region reused for h after S2 consumed u (stream-ordered).
  k_ffn1<<<dim3(INTER / 128, T_TOKENS / 128, NEXP), 256, 0, stream>>>(xb, fc1b, fc1_b, cnt, list, big);
  k_ffn2<<<dim3(HID / 128, T_TOKENS / 128, NEXP), 256, 0, stream>>>(big, fc2b, fc2_b, cnt, list, wl, out);
}
